// Indexer_64175401337409
// MI455X (gfx1250) — 8-bit matrix operand order repaired and checked by reading, compiled, not run on hardware
//
#include <hip/hip_runtime.h>


typedef int   v16i __attribute__((ext_vector_type(16)));
typedef float v8f  __attribute__((ext_vector_type(8)));
typedef float v2f  __attribute__((ext_vector_type(2)));

#define B_  2
#define H_  16
#define S_  4096
#define DM_ 2048
#define DL_ 512
#define TK_ 2048


__device__ __forceinline__ unsigned sw_f32_to_e4m3(float x) {
  unsigned u    = __float_as_uint(x);
  unsigned sign = (u >> 24) & 0x80u;
  unsigned a    = u & 0x7FFFFFFFu;
  int e = (int)(a >> 23) - 127;
  int extra = (-6 - e) > 0 ? (-6 - e) : 0;
  int shift = 20 + extra;  shift = shift > 30 ? 30 : shift;
  unsigned sig  = (a & 0x7FFFFFu) | 0x800000u;
  unsigned r    = sig >> shift;
  unsigned rem  = sig & ((1u << shift) - 1u);
  unsigned half = 1u << (shift - 1);
  r += (rem > half || (rem == half && (r & 1u))) ? 1u : 0u;
  unsigned norm = (r >= 16u) ? ((unsigned)(e + 8) << 3)
                             : (((unsigned)(e + 7) << 3) | (r & 7u));
  norm = norm > 0x7Eu ? 0x7Eu : norm;
  unsigned res = (e < -6) ? r : norm;
  res = (e < -10)            ? 0u    : res;
  res = (a >= 0x43E80000u)   ? 0x7Eu : res;
  res = (a >= 0x7F800000u)   ? 0x7Fu : res;
  return sign | res;
}

__device__ __forceinline__ int pack4_e4m3(float4 f) {
#if __has_builtin(__builtin_amdgcn_cvt_pk_fp8_f32)
  int p = __builtin_amdgcn_cvt_pk_fp8_f32(f.x, f.y, 0, false);
  p     = __builtin_amdgcn_cvt_pk_fp8_f32(f.z, f.w, p, true);
  return p;
#else
  return (int)(sw_f32_to_e4m3(f.x)        | (sw_f32_to_e4m3(f.y) << 8) |
              (sw_f32_to_e4m3(f.z) << 16) | (sw_f32_to_e4m3(f.w) << 24));
#endif
}

__device__ __forceinline__ float fp8_roundtrip(float x) {
#if __has_builtin(__builtin_amdgcn_cvt_pk_fp8_f32) && \
    __has_builtin(__builtin_amdgcn_cvt_f32_fp8)
  int p = __builtin_amdgcn_cvt_pk_fp8_f32(x, x, 0, false);
  return __builtin_amdgcn_cvt_f32_fp8(p, 0);
#else
  unsigned v = sw_f32_to_e4m3(x);
  unsigned e = (v >> 3) & 0xFu;
  unsigned m = v & 7u;
  float r;
  if ((v & 0x7Fu) == 0x7Fu) r = __int_as_float(0x7FC00000);
  else if (e == 0)          r = ldexpf((float)m, -9);
  else                      r = ldexpf((float)(8u + m), (int)e - 10);
  return (v & 0x80u) ? -r : r;
#endif
}

__global__ void idx_proj_fp8(const float* __restrict__ Q,
                             const float* __restrict__ W,
                             const float* __restrict__ bias,
                             float* __restrict__ qlat) {
  const int nTile = blockIdx.x;
  const int mTile = blockIdx.y;
  const int lane  = threadIdx.x;
  const int lo = lane & 15, hi = lane >> 4;

  const int m = mTile * 16 + lo;
  const int n = nTile * 16 + lo;
  const float* qrow = Q + ((size_t)m * S_ + (S_ - 1)) * DM_;
  const float* wrow = W + (size_t)n * DM_;

  v8f acc = {};
  for (int kk = 0; kk < DM_; kk += 128) {
    v16i a, b;
#pragma unroll
    for (int v = 0; v < 16; ++v) {
      int w8 = v & 7;
      int ka = kk + 64 * (v >> 3) + 16 * (w8 >> 1) + 4 * (w8 & 1) + 8 * hi;
      a[v] = pack4_e4m3(*reinterpret_cast<const float4*>(qrow + ka));
      int kb = kk + 64 * (v >> 3) + 16 * (w8 >> 1) + 4 * (w8 & 1) + 8 * hi;
      b[v] = pack4_e4m3(*reinterpret_cast<const float4*>(wrow + kb));
    }
    acc = __builtin_amdgcn_wmma_f32_16x16x128_fp8_fp8(a, b, (short)0, acc, false, false);
  }

  const float bq = fp8_roundtrip(bias[n]);
#pragma unroll
  for (int r = 0; r < 8; ++r) {
    int mr = mTile * 16 + r + 8 * hi;
    qlat[(size_t)mr * DL_ + (nTile * 16 + lo)] = acc[r] + bq;
  }
}

__global__ void idx_scores_f32(const float* __restrict__ qlat,
                               const float* __restrict__ Kd,
                               float* __restrict__ scores) {
  const int nTile = blockIdx.x;
  const int bb    = blockIdx.y;
  const int lane  = threadIdx.x;
  const int lo = lane & 15, hi = lane >> 4;

  const float* arow = qlat + (size_t)(bb * 16 + lo) * DL_;
  const float* brow = Kd + ((size_t)bb * S_ + (size_t)(nTile * 16 + lo)) * DL_;

  v8f acc = {};
  for (int l = 0; l < DL_; l += 4) {
    if ((l & 63) == 0) __builtin_prefetch(brow + l + 128, 0, 0);
    float2 af = *reinterpret_cast<const float2*>(arow + l + 2 * hi);
    float2 bf = *reinterpret_cast<const float2*>(brow + l + 2 * hi);
    v2f a; a[0] = af.x; a[1] = af.y;
    v2f b; b[0] = bf.x; b[1] = bf.y;
    acc = __builtin_amdgcn_wmma_f32_16x16x4_f32(false, a, false, b, (short)0, acc, false, false);
  }
#pragma unroll
  for (int r = 0; r < 8; ++r) {
    int h = r + 8 * hi;
    int s = nTile * 16 + lo;
    float v = acc[r];
    scores[((size_t)(bb * H_ + h)) * S_ + s] = v > 0.f ? v : 0.f;
  }
}

__global__ void idx_topk(const float* __restrict__ scores,
                         float* __restrict__ vals,
                         float* __restrict__ idxf) {
  __shared__ float sv[S_];
  __shared__ int   si[S_];
  const int row = blockIdx.x;
  const float* src = scores + (size_t)row * S_;

  for (int i = threadIdx.x; i < S_; i += blockDim.x) { sv[i] = src[i]; si[i] = i; }
  __syncthreads();

  for (int size = 2; size <= S_; size <<= 1) {
    for (int stride = size >> 1; stride > 0; stride >>= 1) {
      for (int t = threadIdx.x; t < S_ / 2; t += blockDim.x) {
        int i = 2 * t - (t & (stride - 1));
        int j = i + stride;
        bool desc = ((i & size) == 0);
        float vi = sv[i], vj = sv[j];
        int   ii = si[i], jj = si[j];
        bool jBeforeI = (vj > vi) || (vj == vi && jj < ii);
        bool iBeforeJ = (vi > vj) || (vi == vj && ii < jj);
        if (desc ? jBeforeI : iBeforeJ) {
          sv[i] = vj; sv[j] = vi;
          si[i] = jj; si[j] = ii;
        }
      }
      __syncthreads();
    }
  }

  for (int t = threadIdx.x; t < TK_; t += blockDim.x) {
    vals[(size_t)row * TK_ + t] = sv[t];
    idxf[(size_t)row * TK_ + t] = (float)si[t];
  }
}

extern "C" void kernel_launch(void* const* d_in, const int* in_sizes, int n_in,
                              void* d_out, int out_size, void* d_ws, size_t ws_size,
                              hipStream_t stream) {
  (void)in_sizes; (void)n_in; (void)out_size; (void)ws_size;
  const float* Q    = (const float*)d_in[0];
  const float* Kd   = (const float*)d_in[1];
  const float* W    = (const float*)d_in[3];
  const float* bias = (const float*)d_in[4];

  float* out    = (float*)d_out;
  float* scores = out;
  float* tvals  = scores + (size_t)B_ * H_ * S_;
  float* tidx   = tvals  + (size_t)B_ * H_ * TK_;
  float* qlat   = (float*)d_ws;

  idx_proj_fp8 <<<dim3(DL_ / 16, (B_ * H_) / 16), 32, 0, stream>>>(Q, W, bias, qlat);
  idx_scores_f32<<<dim3(S_ / 16, B_),             32, 0, stream>>>(qlat, Kd, scores);
  idx_topk     <<<B_ * H_,                       512, 0, stream>>>(scores, tvals, tidx);
}
